// Net_60559038873657
// MI455X (gfx1250) — matrix operand order repaired and checked by reading, not compiled, not run on hardware
//
#include <hip/hip_runtime.h>
#include <hip/hip_bf16.h>
#include <math.h>

// ---- problem constants (from reference setup) ----
#define NB   16      // batch
#define TT   2048    // seq len
#define BT   32768   // NB*TT
#define HH   128     // hidden
#define G4   512     // 4*H gates
#define DIN0 224     // layer-0 LSTM input dim
#define DAC  192     // conv feature dim
#define D2   32      // x2 dim

typedef __attribute__((ext_vector_type(16))) __bf16 v16bf;
typedef __attribute__((ext_vector_type(8)))  __bf16 v8bf;
typedef __attribute__((ext_vector_type(8)))  float  v8f;

static __device__ __forceinline__ unsigned short f2bf(float f) {
  unsigned int u = __float_as_uint(f);
  u += 0x7fffu + ((u >> 16) & 1u);           // round-to-nearest-even
  return (unsigned short)(u >> 16);
}
static __device__ __forceinline__ float sigf(float x) { return 1.0f / (1.0f + __expf(-x)); }

// CDNA5 async global->LDS copy (ASYNCcnt-tracked). LDS dest = low 32 bits of the
// generic shared pointer (flat LDS aperture keeps the offset in addr[31:0]).
static __device__ __forceinline__ void async_g2l_b128(unsigned lds_off, const void* gaddr) {
  asm volatile("global_load_async_to_lds_b128 %0, %1, off"
               :: "v"(lds_off), "v"(gaddr) : "memory");
}
static __device__ __forceinline__ void wait_async_le8() {
  asm volatile("s_wait_asynccnt 8" ::: "memory");
}
static __device__ __forceinline__ void wait_async_0() {
  asm volatile("s_wait_asynccnt 0" ::: "memory");
}

// A-tile 16x32 bf16: lane L holds row M=L&15; K = kb..kb+15, kb = k0 + 16*(L>=16)
static __device__ __forceinline__ v16bf load_a16x32(const unsigned short* rowbase, int k0, int lane) {
  int kb = k0 + ((lane >> 4) << 4);
  const __bf16* p = reinterpret_cast<const __bf16*>(rowbase + kb);
  v8bf lo = *reinterpret_cast<const v8bf*>(p);
  v8bf hi = *reinterpret_cast<const v8bf*>(p + 8);
  v16bf a;
#pragma unroll
  for (int i = 0; i < 8; ++i) { a[i] = lo[i]; a[i + 8] = hi[i]; }
  return a;
}
// B-tile 32x16 bf16 (B = W^T, W row-major [N][K]): lane L holds col N=n0+(L&15),
// K = k0 + 16*(L>=16) .. +15  -> one contiguous 32B load per lane
static __device__ __forceinline__ v16bf load_b32x16(const unsigned short* w, int ldw, int n0, int k0, int lane) {
  int row = n0 + (lane & 15);
  int kb = k0 + ((lane >> 4) << 4);
  return *reinterpret_cast<const v16bf*>(w + (size_t)row * ldw + kb);
}

// =============== kernel 1: conv1+bn+relu+pool, conv2+bn+relu+pool, concat x2 -> feat(bf16) ===============
__global__ __launch_bounds__(256) void conv_feat_kernel(
    const float* __restrict__ x1, const float* __restrict__ x2,
    const float* __restrict__ w1, const float* __restrict__ cb1,
    const float* __restrict__ g1, const float* __restrict__ be1,
    const float* __restrict__ m1, const float* __restrict__ v1,
    const float* __restrict__ w2, const float* __restrict__ cb2,
    const float* __restrict__ g2, const float* __restrict__ be2,
    const float* __restrict__ m2, const float* __restrict__ v2,
    unsigned short* __restrict__ feat) {
  int gtid = blockIdx.x * 256 + threadIdx.x;
  int sample = gtid >> 5;                 // b*T + t
  int lane = gtid & 31;
  const float* xs = x1 + (size_t)sample * 192;   // [3][64]

  float a1c[6], s1c[6];
#pragma unroll
  for (int c = 0; c < 6; ++c) {
    float sc = g1[c] * rsqrtf(v1[c] + 1e-5f);
    a1c[c] = sc; s1c[c] = (cb1[c] - m1[c]) * sc + be1[c];
  }
  // conv1 (pad1) + bn + relu at positions 2*lane, 2*lane+1, maxpool2 -> pooled[c] at pos=lane
  float pooled[6];
#pragma unroll
  for (int c = 0; c < 6; ++c) {
    float best = -INFINITY;
#pragma unroll
    for (int q = 0; q < 2; ++q) {
      int p = 2 * lane + q;
      float acc = 0.f;
#pragma unroll
      for (int ic = 0; ic < 3; ++ic) {
        const float* xi = xs + ic * 64;
        const float* wc = w1 + (c * 3 + ic) * 3;
        float xm1 = (p > 0)  ? xi[p - 1] : 0.f;
        float x0  = xi[p];
        float xp1 = (p < 63) ? xi[p + 1] : 0.f;
        acc += wc[0] * xm1 + wc[1] * x0 + wc[2] * xp1;
      }
      best = fmaxf(best, fmaxf(acc * a1c[c] + s1c[c], 0.f));
    }
    pooled[c] = best;
  }
  // neighbors via shuffle (seq len 32 == wave32)
  float left[6], right[6];
#pragma unroll
  for (int c = 0; c < 6; ++c) {
    float lv = __shfl(pooled[c], lane - 1, 32);
    float rv = __shfl(pooled[c], lane + 1, 32);
    left[c]  = (lane == 0)  ? 0.f : lv;
    right[c] = (lane == 31) ? 0.f : rv;
  }
  float a2c[12], s2c[12];
#pragma unroll
  for (int c = 0; c < 12; ++c) {
    float sc = g2[c] * rsqrtf(v2[c] + 1e-5f);
    a2c[c] = sc; s2c[c] = (cb2[c] - m2[c]) * sc + be2[c];
  }
  float y2[12];
#pragma unroll
  for (int oc = 0; oc < 12; ++oc) {
    float acc = 0.f;
#pragma unroll
    for (int c = 0; c < 6; ++c) {
      const float* wc = w2 + (oc * 6 + c) * 3;
      acc += wc[0] * left[c] + wc[1] * pooled[c] + wc[2] * right[c];
    }
    y2[oc] = fmaxf(acc * a2c[oc] + s2c[oc], 0.f);
  }
  // maxpool2 over positions -> lanes 0..15 hold final 16 positions; write bf16 feat
  unsigned short* frow = feat + (size_t)sample * DIN0;
#pragma unroll
  for (int oc = 0; oc < 12; ++oc) {
    float va = __shfl(y2[oc], 2 * (lane & 15), 32);
    float vb = __shfl(y2[oc], 2 * (lane & 15) + 1, 32);
    if (lane < 16) frow[oc * 16 + lane] = f2bf(fmaxf(va, vb));
  }
  frow[DAC + lane] = f2bf(x2[(size_t)sample * D2 + lane]);
}

// =============== small conversion helpers ===============
__global__ void cvt_bf16_kernel(const float* __restrict__ s, unsigned short* __restrict__ d, int n) {
  int i = blockIdx.x * 256 + threadIdx.x;
  if (i < n) d[i] = f2bf(s[i]);
}
__global__ void bias_add_kernel(const float* __restrict__ a, const float* __restrict__ b,
                                float* __restrict__ o, int n) {
  int i = blockIdx.x * 256 + threadIdx.x;
  if (i < n) o[i] = a[i] + b[i];
}

// =============== kernel 2: input projection GEMM  xg = A @ Wih^T + (bih+bhh)  (bf16 WMMA) ===============
// grid (BT/128, 8), block 256 (8 waves). Wave w: rows m0=128*bx+16*w, cols 64*by..+63 (4 N-tiles).
__global__ __launch_bounds__(256) void proj_kernel(
    const unsigned short* __restrict__ A, int lda, int KD,
    const unsigned short* __restrict__ Wb, const float* __restrict__ bias,
    float* __restrict__ xg, const int* __restrict__ length, int gather) {
  const int lane = threadIdx.x & 31;
  const int wave = threadIdx.x >> 5;
  const int m0 = blockIdx.x * 128 + wave * 16;
  const int n0 = blockIdx.y * 64;

  int r = m0 + (lane & 15);
  int row_src = r;
  if (gather) {                       // _reverse_padded on the time axis (per sample)
    int b = r >> 11, t = r & (TT - 1);
    int len = length[b];
    int ts = (t < len) ? (len - 1 - t) : t;
    row_src = (b << 11) + ts;
  }
  const unsigned short* arow = A + (size_t)row_src * lda;

  v8f acc[4] = {};
  for (int k0 = 0; k0 < KD; k0 += 32) {
    v16bf a = load_a16x32(arow, k0, lane);
#pragma unroll
    for (int j = 0; j < 4; ++j) {
      v16bf bm = load_b32x16(Wb, KD, n0 + 16 * j, k0, lane);
      acc[j] = __builtin_amdgcn_wmma_f32_16x16x32_bf16(false, a, false, bm,
                                                       (short)0, acc[j], false, false);
    }
  }
#pragma unroll
  for (int j = 0; j < 4; ++j) {
    int n = n0 + 16 * j + (lane & 15);
    float bv = bias[n];
#pragma unroll
    for (int e = 0; e < 8; ++e) {
      int m = m0 + e + 8 * (lane >> 4);
      xg[(size_t)m * G4 + n] = acc[j][e] + bv;
    }
  }
}

// =============== kernel 3: recurrent LSTM scan (one workgroup per direction) ===============
// LDS: Whh (512x128 bf16) + h (16x128 bf16) + c (16x128 f32) + double-buffered xg_t (2x 16x512 f32).
// xg[:,t+1,:] is prefetched with GLOBAL_LOAD_ASYNC_TO_LDS_B128 while step t computes.
__global__ __launch_bounds__(256) void lstm_kernel(
    const float* __restrict__ xg_all, unsigned long long xg_dstride,
    const unsigned short* __restrict__ whh_all, int layer,
    unsigned short* __restrict__ h0_all, unsigned long long h0_dstride,
    float* __restrict__ outbuf, const int* __restrict__ length) {
  extern __shared__ char smem[];
  unsigned short* whhL = (unsigned short*)smem;           // 512*128 bf16
  unsigned short* hL   = whhL + G4 * HH;                  // 16*128 bf16
  float* cL  = (float*)(hL + NB * HH);                    // 16*128 f32
  float* xgA = cL + NB * HH;                              // 16*512 f32 (buffer 0)
  float* xgB = xgA + NB * G4;                             // 16*512 f32 (buffer 1)

  const int d = blockIdx.x;
  const int tid = threadIdx.x, lane = tid & 31, wave = tid >> 5;
  const float* xg = xg_all + (size_t)d * xg_dstride;
  const unsigned short* whh = whh_all + (size_t)(d * 2 + layer) * G4 * HH;
  unsigned short* h0p = h0_all ? (h0_all + (size_t)d * h0_dstride) : nullptr;

  const unsigned ldsbase[2] = { (unsigned)(size_t)(void*)xgA, (unsigned)(size_t)(void*)xgB };

  for (int i = tid; i < G4 * HH / 8; i += 256)            // 8192 uint4
    ((uint4*)whhL)[i] = ((const uint4*)whh)[i];
  for (int i = tid; i < NB * HH; i += 256) { hL[i] = 0; cL[i] = 0.f; }

  // prefetch xg[:,0,:] into buffer 0 (8 async b128 per lane; chunk c covers floats 4c..4c+3
  // of the row-major [16][512] slice, matching the m*512+col read layout)
#pragma unroll
  for (int it = 0; it < 8; ++it) {
    int c = tid + 256 * it;
    int b = c >> 7, col4 = (c & 127) << 2;
    async_g2l_b128(ldsbase[0] + (unsigned)c * 16, xg + ((size_t)b * TT) * G4 + col4);
  }

  const int j0 = wave * 16;                               // this wave's hidden-unit tile
  const unsigned short* harow = hL + (size_t)(lane & 15) * HH;

  for (int t = 0; t < TT; ++t) {
    const int cur = t & 1;
    const bool more = (t + 1) < TT;
    if (more) {                                           // prefetch next step's slice
#pragma unroll
      for (int it = 0; it < 8; ++it) {
        int c = tid + 256 * it;
        int b = c >> 7, col4 = (c & 127) << 2;
        async_g2l_b128(ldsbase[cur ^ 1] + (unsigned)c * 16,
                       xg + ((size_t)b * TT + (t + 1)) * G4 + col4);
      }
      wait_async_le8();   // async loads retire in order -> current slice complete
    } else {
      wait_async_0();
    }
    __syncthreads();      // current xg slice + previous h update visible to all waves

    const float* xgL = cur ? xgB : xgA;
    v8f acc[4] = {};
#pragma unroll
    for (int k = 0; k < 4; ++k) {
      v16bf a = load_a16x32(harow, k * 32, lane);
#pragma unroll
      for (int g = 0; g < 4; ++g) {                       // gates i,f,g,o
        v16bf bm = load_b32x16(whhL, HH, g * HH + j0, k * 32, lane);
        acc[g] = __builtin_amdgcn_wmma_f32_16x16x32_bf16(false, a, false, bm,
                                                         (short)0, acc[g], false, false);
      }
    }
    __syncthreads();                                      // all reads of hL complete before h is overwritten

    int n = j0 + (lane & 15);
#pragma unroll
    for (int e = 0; e < 8; ++e) {
      int m = e + 8 * (lane >> 4);                        // batch row
      float gi = acc[0][e] + xgL[m * G4 + 0 * HH + n];
      float gf = acc[1][e] + xgL[m * G4 + 1 * HH + n];
      float gg = acc[2][e] + xgL[m * G4 + 2 * HH + n];
      float go = acc[3][e] + xgL[m * G4 + 3 * HH + n];
      float c  = sigf(gf) * cL[m * HH + n] + sigf(gi) * tanhf(gg);
      float h  = sigf(go) * tanhf(c);
      cL[m * HH + n] = c;
      hL[m * HH + n] = f2bf(h);
      if (layer == 0) {
        h0p[((size_t)m * TT + t) * HH + n] = f2bf(h);     // unmasked inter-layer output
      } else {
        int len = length[m];
        size_t tdst; float val;
        if (t < len) { tdst = (d == 0) ? (size_t)t : (size_t)(len - 1 - t); val = h; }
        else         { tdst = (size_t)t; val = 0.f; }     // mask * un-reverse
        outbuf[((size_t)m * TT + tdst) * 256 + (size_t)d * HH + n] = val;
      }
    }
    __syncthreads();
  }
}

// =============== kernel 4: masked softmax attention pooling + output projection ===============
__global__ __launch_bounds__(256) void attn_kernel(
    const float* __restrict__ outb, const float* __restrict__ attn_w,
    const float* __restrict__ W2, const float* __restrict__ b2,
    const int* __restrict__ length, float* __restrict__ d_out) {
  __shared__ float pot[TT];
  __shared__ float red[256];
  __shared__ float aw[256];
  __shared__ float w2s[4 * 256];
  __shared__ float b2s[4];
  int b = blockIdx.x, tid = threadIdx.x;
  int len = length[b];
  const float* ob = outb + (size_t)b * TT * 256;

  aw[tid] = attn_w[tid];
  for (int i = tid; i < 1024; i += 256) w2s[i] = W2[i];
  if (tid < 4) b2s[tid] = b2[tid];
  __syncthreads();

  for (int t = tid; t < TT; t += 256) {
    const float* row = ob + (size_t)t * 256;
    float s = 0.f;
    for (int h = 0; h < 256; ++h) s += row[h] * aw[h];
    pot[t] = s;
  }
  __syncthreads();
  float mx = -INFINITY;
  for (int t = tid; t < TT; t += 256) if (t < len) mx = fmaxf(mx, pot[t]);
  red[tid] = mx; __syncthreads();
  for (int s = 128; s > 0; s >>= 1) { if (tid < s) red[tid] = fmaxf(red[tid], red[tid + s]); __syncthreads(); }
  mx = red[0]; __syncthreads();
  float sm = 0.f;
  for (int t = tid; t < TT; t += 256) {
    float e = (t < len) ? __expf(pot[t] - mx) : 0.f;
    pot[t] = e; sm += e;
  }
  red[tid] = sm; __syncthreads();
  for (int s = 128; s > 0; s >>= 1) { if (tid < s) red[tid] += red[tid + s]; __syncthreads(); }
  float inv = 1.f / red[0];
  __syncthreads();
  {   // out_final: thread == hidden index
    float acc = 0.f;
    for (int t = 0; t < TT; ++t) acc += pot[t] * ob[(size_t)t * 256 + tid];
    d_out[(size_t)NB * TT * 4 + (size_t)b * 256 + tid] = acc * inv;
  }
  for (int t = tid; t < TT; t += 256) {   // logits2 (B,T,4)
    const float* row = ob + (size_t)t * 256;
    float s0 = 0, s1 = 0, s2 = 0, s3 = 0;
    for (int h = 0; h < 256; ++h) {
      float v = row[h];
      s0 += v * w2s[h]; s1 += v * w2s[256 + h]; s2 += v * w2s[512 + h]; s3 += v * w2s[768 + h];
    }
    float* o = d_out + ((size_t)b * TT + t) * 4;
    o[0] = s0 + b2s[0]; o[1] = s1 + b2s[1]; o[2] = s2 + b2s[2]; o[3] = s3 + b2s[3];
  }
}

// =============== host orchestration ===============
extern "C" void kernel_launch(void* const* d_in, const int* in_sizes, int n_in,
                              void* d_out, int out_size, void* d_ws, size_t ws_size,
                              hipStream_t stream) {
  (void)in_sizes; (void)n_in; (void)out_size; (void)ws_size;
  const float* x1 = (const float*)d_in[0];
  const float* x2 = (const float*)d_in[1];
  const int*   length = (const int*)d_in[2];
  const float* conv1_w = (const float*)d_in[3];  const float* conv1_b = (const float*)d_in[4];
  const float* bn1_g = (const float*)d_in[5];    const float* bn1_b = (const float*)d_in[6];
  const float* bn1_m = (const float*)d_in[7];    const float* bn1_v = (const float*)d_in[8];
  const float* conv2_w = (const float*)d_in[9];  const float* conv2_b = (const float*)d_in[10];
  const float* bn2_g = (const float*)d_in[11];   const float* bn2_b = (const float*)d_in[12];
  const float* bn2_m = (const float*)d_in[13];   const float* bn2_v = (const float*)d_in[14];
  const float* attn_w = (const float*)d_in[15];
  const float* W2 = (const float*)d_in[16];      const float* b2 = (const float*)d_in[17];
  const float* Wih[4]; const float* Whh[4]; const float* bih[4]; const float* bhh[4];
  for (int d = 0; d < 2; ++d)
    for (int l = 0; l < 2; ++l) {
      int base = 18 + (d * 2 + l) * 4;
      int idx = d * 2 + l;
      Wih[idx] = (const float*)d_in[base + 0];
      Whh[idx] = (const float*)d_in[base + 1];
      bih[idx] = (const float*)d_in[base + 2];
      bhh[idx] = (const float*)d_in[base + 3];
    }

  char* ws = (char*)d_ws;
  size_t off = 0;
  auto alloc = [&](size_t bytes) { size_t p = off; off += (bytes + 255) & ~(size_t)255; return p; };
  unsigned short* feat  = (unsigned short*)(ws + alloc((size_t)BT * DIN0 * 2));
  unsigned short* wihb  = (unsigned short*)(ws + alloc((size_t)4 * G4 * DIN0 * 2));  // uniform slots
  unsigned short* whhb  = (unsigned short*)(ws + alloc((size_t)4 * G4 * HH * 2));
  float*          biasb = (float*)(ws + alloc((size_t)4 * G4 * 4));
  float*          xg    = (float*)(ws + alloc((size_t)2 * BT * G4 * 4));
  unsigned short* h0    = (unsigned short*)(ws + alloc((size_t)2 * BT * HH * 2));
  float*          outb  = (float*)(ws + alloc((size_t)BT * 256 * 4));

  conv_feat_kernel<<<BT * 32 / 256, 256, 0, stream>>>(
      x1, x2, conv1_w, conv1_b, bn1_g, bn1_b, bn1_m, bn1_v,
      conv2_w, conv2_b, bn2_g, bn2_b, bn2_m, bn2_v, feat);

  const int din[2] = {DIN0, HH};
  for (int d = 0; d < 2; ++d)
    for (int l = 0; l < 2; ++l) {
      int idx = d * 2 + l;
      int nW = G4 * din[l];
      cvt_bf16_kernel<<<(nW + 255) / 256, 256, 0, stream>>>(Wih[idx], wihb + (size_t)idx * G4 * DIN0, nW);
      cvt_bf16_kernel<<<(G4 * HH + 255) / 256, 256, 0, stream>>>(Whh[idx], whhb + (size_t)idx * G4 * HH, G4 * HH);
      bias_add_kernel<<<2, 256, 0, stream>>>(bih[idx], bhh[idx], biasb + (size_t)idx * G4, G4);
    }

  dim3 pg(BT / 128, 8);
  size_t lstm_lds = (size_t)G4 * HH * 2 + (size_t)NB * HH * 2 + (size_t)NB * HH * 4
                  + (size_t)2 * NB * G4 * 4;   // Whh + h + c + double-buffered xg slice

  // layer 0: proj (d=1 gathers reversed time), then recurrent scan for both dirs
  for (int d = 0; d < 2; ++d)
    proj_kernel<<<pg, 256, 0, stream>>>(feat, DIN0, DIN0,
                                        wihb + (size_t)(d * 2) * G4 * DIN0, biasb + (size_t)(d * 2) * G4,
                                        xg + (size_t)d * BT * G4, length, d == 1 ? 1 : 0);
  lstm_kernel<<<2, 256, lstm_lds, stream>>>(xg, (unsigned long long)BT * G4, whhb, 0,
                                            h0, (unsigned long long)BT * HH, nullptr, length);
  // layer 1
  for (int d = 0; d < 2; ++d)
    proj_kernel<<<pg, 256, 0, stream>>>(h0 + (size_t)d * BT * HH, HH, HH,
                                        wihb + (size_t)(d * 2 + 1) * G4 * DIN0, biasb + (size_t)(d * 2 + 1) * G4,
                                        xg + (size_t)d * BT * G4, length, 0);
  lstm_kernel<<<2, 256, lstm_lds, stream>>>(xg, (unsigned long long)BT * G4, whhb, 1,
                                            nullptr, 0, outb, length);

  attn_kernel<<<NB, 256, 0, stream>>>(outb, attn_w, W2, b2, length, (float*)d_out);
}
